// GNN_20813411516770
// MI455X (gfx1250) — matrix operand order repaired and checked by reading, not compiled, not run on hardware
//
#include <hip/hip_runtime.h>
#include <hip/hip_bf16.h>

#define BN_EPS 0.001f

typedef __attribute__((ext_vector_type(16))) _Float16 v16h;
typedef __attribute__((ext_vector_type(8)))  _Float16 v8h;
typedef __attribute__((ext_vector_type(8)))  float    v8f;

__device__ __forceinline__ float gelu_exact(float x) {
  return 0.5f * x * (1.0f + erff(x * 0.70710678118654752f));
}

// ---------------- elementwise: optional BN then f32 -> f16 convert (row-major) ----------------
__global__ __launch_bounds__(256) void k_bn_to_h(
    const float* __restrict__ X,
    const float* __restrict__ g, const float* __restrict__ b,
    const float* __restrict__ mu, const float* __restrict__ var,
    _Float16* __restrict__ out, int total, int D, int useBN)
{
  int tid = blockIdx.x * 256 + threadIdx.x;
  if (tid >= total) return;
  float v = X[tid];
  if (useBN) {
    int k = tid % D;
    float s = g[k] * rsqrtf(var[k] + BN_EPS);
    v = (v - mu[k]) * s + b[k];
  }
  out[tid] = (_Float16)v;
}

// ---------------- weight convert + transpose: W(D x 256) f32 -> WT(256 x D) f16 ----------------
__global__ __launch_bounds__(256) void k_cvt_wT(
    const float* __restrict__ W, _Float16* __restrict__ WT, int D)
{
  int tid = blockIdx.x * 256 + threadIdx.x;    // over D*256
  int k = tid >> 8;
  int n = tid & 255;
  WT[(size_t)n * D + k] = (_Float16)W[tid];
}

// ---------------- concat [BN(x) | BN(agg)] -> f16 (4096 x 512) ----------------
__global__ __launch_bounds__(256) void k_concat2h(
    const float* __restrict__ x, const float* __restrict__ agg,
    const float* __restrict__ g, const float* __restrict__ b,
    const float* __restrict__ mu, const float* __restrict__ var,
    _Float16* __restrict__ U)
{
  int tid = blockIdx.x * 256 + threadIdx.x;   // over 4096*512
  int row = tid >> 9;
  int c = tid & 511;
  float v = (c < 256) ? x[row * 256 + c] : agg[row * 256 + (c - 256)];
  float s = g[c] * rsqrtf(var[c] + BN_EPS);
  v = (v - mu[c]) * s + b[c];
  U[tid] = (_Float16)v;
}

// ---------------- WMMA GEMM: Y(R x 256) = act( Xh(R x D) @ W + bias/Cper ) [+ resid] ----------
// WT is the f16 weight matrix stored TRANSPOSED: WT[n][k], n in [0,256), k in [0,D).
// Block = 256 threads = 8 waves; wave (rt,ct) computes rows [rowBase,+16) x cols [ct*64,+64)
// via 4 accumulators sharing one A fragment. No LDS, no barriers: A and B fragments are
// contiguous 16-half runs -> pairs of global b128 loads feeding v_wmma directly.
__global__ __launch_bounds__(256) void k_wmma_gemm(
    const _Float16* __restrict__ Xh,
    const _Float16* __restrict__ WT,
    const float* __restrict__ bias,    // 256 or null
    const float* __restrict__ Cper,    // Bx256 per-batch row (b = row>>6) or null
    const float* __restrict__ resid,   // R x 256 or null
    float* __restrict__ Y,
    int D, int doGelu)
{
  const int lane   = threadIdx.x & 31;
  const int wave   = threadIdx.x >> 5;
  const int rt     = wave & 1;
  const int ct     = wave >> 1;                  // 0..3
  const int rowBase = blockIdx.x * 32 + rt * 16;
  const int nLoc   = lane & 15;
  const int halfId = lane >> 4;

  // A: lane holds row M = lane%16; halves i=0..15 -> K = halfId*16 + i
  const _Float16* xrow = Xh + (size_t)(rowBase + nLoc) * D;
  // B: lane holds column n; halves i=0..15 -> K = halfId*16 + i (contiguous in WT row n)
  const _Float16* bc0 = WT + (size_t)(ct * 64 +  0 + nLoc) * D + halfId * 16;
  const _Float16* bc1 = WT + (size_t)(ct * 64 + 16 + nLoc) * D + halfId * 16;
  const _Float16* bc2 = WT + (size_t)(ct * 64 + 32 + nLoc) * D + halfId * 16;
  const _Float16* bc3 = WT + (size_t)(ct * 64 + 48 + nLoc) * D + halfId * 16;

  v8f acc0 = {}, acc1 = {}, acc2 = {}, acc3 = {};

  for (int k0 = 0; k0 < D; k0 += 32) {
    v16h a;
    {
      v8h alo = *(const v8h*)(xrow + k0 + halfId * 16);
      v8h ahi = *(const v8h*)(xrow + k0 + halfId * 16 + 8);
#pragma unroll
      for (int i = 0; i < 8; ++i) { a[i] = alo[i]; a[8 + i] = ahi[i]; }
    }
    v16h b0, b1, b2, b3;
    {
      v8h t0 = *(const v8h*)(bc0 + k0), t1 = *(const v8h*)(bc0 + k0 + 8);
      v8h u0 = *(const v8h*)(bc1 + k0), u1 = *(const v8h*)(bc1 + k0 + 8);
      v8h v0 = *(const v8h*)(bc2 + k0), v1 = *(const v8h*)(bc2 + k0 + 8);
      v8h w0 = *(const v8h*)(bc3 + k0), w1 = *(const v8h*)(bc3 + k0 + 8);
#pragma unroll
      for (int i = 0; i < 8; ++i) {
        b0[i] = t0[i]; b0[8 + i] = t1[i];
        b1[i] = u0[i]; b1[8 + i] = u1[i];
        b2[i] = v0[i]; b2[8 + i] = v1[i];
        b3[i] = w0[i]; b3[8 + i] = w1[i];
      }
    }
    acc0 = __builtin_amdgcn_wmma_f32_16x16x32_f16(false, a, false, b0, (short)0, acc0, false, false);
    acc1 = __builtin_amdgcn_wmma_f32_16x16x32_f16(false, a, false, b1, (short)0, acc1, false, false);
    acc2 = __builtin_amdgcn_wmma_f32_16x16x32_f16(false, a, false, b2, (short)0, acc2, false, false);
    acc3 = __builtin_amdgcn_wmma_f32_16x16x32_f16(false, a, false, b3, (short)0, acc3, false, false);
  }

  // epilogue: C/D layout -> lane covers col n, rows M = halfId*8 + v
  v8f accs[4] = {acc0, acc1, acc2, acc3};
#pragma unroll
  for (int c = 0; c < 4; ++c) {
    const int n = ct * 64 + c * 16 + nLoc;
    const float bv = bias ? bias[n] : 0.0f;
#pragma unroll
    for (int v = 0; v < 8; ++v) {
      int m = rowBase + halfId * 8 + v;
      float val = accs[c][v] + bv;
      if (Cper)  val += Cper[(m >> 6) * 256 + n];
      if (doGelu) val = gelu_exact(val);
      if (resid) val += resid[(size_t)m * 256 + n];
      Y[(size_t)m * 256 + n] = val;
    }
  }
}

// ---------------- time FFN layer0 (in_dim = 1) ----------------
__global__ __launch_bounds__(256) void k_te0(
    const float* __restrict__ tm,
    const float* __restrict__ g, const float* __restrict__ b,
    const float* __restrict__ mu, const float* __restrict__ var,
    const float* __restrict__ W, const float* __restrict__ bias,
    float* __restrict__ out)
{
  int tid = blockIdx.x * 256 + threadIdx.x;        // 64*256
  int bb = tid >> 8, j = tid & 255;
  float s = g[0] * rsqrtf(var[0] + BN_EPS);
  float xn = (tm[bb] - mu[0]) * s + b[0];
  out[tid] = gelu_exact(xn * W[j] + bias[j]);
}

// ---------------- pre FFN layer0 (in_dim = 4) ----------------
__global__ __launch_bounds__(256) void k_pre0(
    const float* __restrict__ p,
    const float* __restrict__ g, const float* __restrict__ b,
    const float* __restrict__ mu, const float* __restrict__ var,
    const float* __restrict__ W, const float* __restrict__ bias,
    float* __restrict__ out)
{
  int tid = blockIdx.x * 256 + threadIdx.x;        // 4096*256
  int row = tid >> 8, j = tid & 255;
  float acc = bias[j];
#pragma unroll
  for (int k = 0; k < 4; ++k) {
    float xn = (p[row * 4 + k] - mu[k]) * rsqrtf(var[k] + BN_EPS) * g[k] + b[k];
    acc += xn * W[k * 256 + j];
  }
  out[tid] = gelu_exact(acc);
}

// ---------------- build adjacency-count matrix A (64x64) + 1/max(cnt,1) ----------------
__global__ __launch_bounds__(256) void k_buildA(
    const int* __restrict__ edges, int E,
    float* __restrict__ A, float* __restrict__ cntInv)
{
  __shared__ int cnt[4096];
  int t = threadIdx.x;
  for (int i = t; i < 4096; i += 256) cnt[i] = 0;
  __syncthreads();
  for (int e = t; e < E; e += 256) {
    int r = edges[2 * e], c = edges[2 * e + 1];
    atomicAdd(&cnt[r * 64 + c], 1);
  }
  __syncthreads();
  for (int i = t; i < 4096; i += 256) A[i] = (float)cnt[i];
  __syncthreads();
  if (t < 64) {
    int s = 0;
    for (int j = 0; j < 64; ++j) s += cnt[t * 64 + j];
    cntInv[t] = 1.0f / fmaxf((float)s, 1.0f);
  }
}

// ---------------- agg[b,n,:] = cntInv[n] * sum_j A[n,j] * M1[b,j,:] ----------------
__global__ __launch_bounds__(256) void k_agg(
    const float* __restrict__ A, const float* __restrict__ cntInv,
    const float* __restrict__ M1, float* __restrict__ out)
{
  int tid = blockIdx.x * 256 + threadIdx.x;        // 4096*256
  int h = tid & 255;
  int row = tid >> 8;
  int bb = row >> 6, n = row & 63;
  float acc = 0.0f;
#pragma unroll 8
  for (int j = 0; j < 64; ++j)
    acc += A[n * 64 + j] * M1[(size_t)((bb << 6) + j) * 256 + h];
  out[(size_t)row * 256 + h] = acc * cntInv[n];
}

// ---------------- final linear: out(4096 x 4) = Xp @ Wl + bl ----------------
__global__ __launch_bounds__(256) void k_final(
    const float* __restrict__ Xp, const float* __restrict__ Wl,
    const float* __restrict__ bl, float* __restrict__ out)
{
  int tid = blockIdx.x * 256 + threadIdx.x;        // 4096*4
  int row = tid >> 2, f = tid & 3;
  float acc = bl[f];
  for (int k = 0; k < 256; ++k) acc += Xp[(size_t)row * 256 + k] * Wl[k * 4 + f];
  out[tid] = acc;
}

// =================================================================================
struct Lay { const float *g, *b, *mu, *var, *W, *bias; };
static inline Lay getLay(void* const* d_in, int base) {
  return Lay{ (const float*)d_in[base + 0], (const float*)d_in[base + 1],
              (const float*)d_in[base + 2], (const float*)d_in[base + 3],
              (const float*)d_in[base + 4], (const float*)d_in[base + 5] };
}

static void bnToH(const float* X, const Lay& L, int off, _Float16* out,
                  int total, int D, hipStream_t s) {
  k_bn_to_h<<<(total + 255) / 256, 256, 0, s>>>(X, L.g + off, L.b + off,
                                                L.mu + off, L.var + off,
                                                out, total, D, 1);
}
static void cvtWT(const float* W, _Float16* out, int D, hipStream_t s) {
  k_cvt_wT<<<(D * 256 + 255) / 256, 256, 0, s>>>(W, out, D);
}
static void gemm(const _Float16* Xh, const _Float16* WT, const float* bias,
                 const float* Cper, const float* resid, float* Y,
                 int R, int D, int doGelu, hipStream_t s) {
  k_wmma_gemm<<<R / 32, 256, 0, s>>>(Xh, WT, bias, Cper, resid, Y, D, doGelu);
}

static void graph_conv(const Lay& p0, const Lay& p1, const Lay& u0, const Lay& u1,
                       const float* xin, float* m0buf, float* m1buf, float* aggbuf,
                       float* xout, const float* te, _Float16* teh, float* cte,
                       _Float16* Xh, _Float16* Wh, _Float16* Wh2,
                       const float* A, const float* cntInv, hipStream_t s)
{
  // prep layer0, te slab -> per-batch constant (includes bias)
  k_bn_to_h<<<64, 256, 0, s>>>(te, p0.g + 256, p0.b + 256, p0.mu + 256, p0.var + 256,
                               teh, 64 * 256, 256, 1);
  cvtWT(p0.W + 256 * 256, Wh2, 256, s);
  gemm(teh, Wh2, p0.bias, nullptr, nullptr, cte, 64, 256, 0, s);
  // prep layer0, node slab
  bnToH(xin, p0, 0, Xh, 4096 * 256, 256, s);
  cvtWT(p0.W, Wh, 256, s);
  gemm(Xh, Wh, nullptr, cte, nullptr, m0buf, 4096, 256, 1, s);
  // prep layer1 -> M1
  bnToH(m0buf, p1, 0, Xh, 4096 * 256, 256, s);
  cvtWT(p1.W, Wh, 256, s);
  gemm(Xh, Wh, p1.bias, nullptr, nullptr, m1buf, 4096, 256, 1, s);
  // aggregate (segment mean == count-matrix matmul)
  k_agg<<<4096, 256, 0, s>>>(A, cntInv, m1buf, aggbuf);
  // upd layer0, te slab -> per-batch constant
  k_bn_to_h<<<64, 256, 0, s>>>(te, u0.g + 512, u0.b + 512, u0.mu + 512, u0.var + 512,
                               teh, 64 * 256, 256, 1);
  cvtWT(u0.W + 512 * 256, Wh2, 256, s);
  gemm(teh, Wh2, u0.bias, nullptr, nullptr, cte, 64, 256, 0, s);
  // upd layer0, concat [x | agg] (D = 512)
  k_concat2h<<<(4096 * 512) / 256, 256, 0, s>>>(xin, aggbuf, u0.g, u0.b, u0.mu, u0.var, Xh);
  cvtWT(u0.W, Wh, 512, s);  // rows 0..511 of W (768x256) -> WT 256x512
  gemm(Xh, Wh, nullptr, cte, nullptr, m0buf, 4096, 512, 1, s);
  // upd layer1 + residual
  bnToH(m0buf, u1, 0, Xh, 4096 * 256, 256, s);
  cvtWT(u1.W, Wh, 256, s);
  gemm(Xh, Wh, u1.bias, nullptr, xin, xout, 4096, 256, 1, s);
}

extern "C" void kernel_launch(void* const* d_in, const int* in_sizes, int n_in,
                              void* d_out, int out_size, void* d_ws, size_t ws_size,
                              hipStream_t stream)
{
  (void)n_in; (void)out_size; (void)ws_size;
  const float* p     = (const float*)d_in[0];
  const float* tm    = (const float*)d_in[1];
  const int*   edges = (const int*)d_in[2];
  const int E = in_sizes[2] / 2;

  Lay t0  = getLay(d_in, 3),  t1  = getLay(d_in, 9);
  Lay pr0 = getLay(d_in, 15), pr1 = getLay(d_in, 21);
  Lay c1p0 = getLay(d_in, 27), c1p1 = getLay(d_in, 33);
  Lay c1u0 = getLay(d_in, 39), c1u1 = getLay(d_in, 45);
  Lay c2p0 = getLay(d_in, 51), c2p1 = getLay(d_in, 57);
  Lay c2u0 = getLay(d_in, 63), c2u1 = getLay(d_in, 69);
  Lay po0 = getLay(d_in, 75), po1 = getLay(d_in, 81);
  const float* Wl = (const float*)d_in[87];
  const float* bl = (const float*)d_in[88];

  // workspace carve (~21 MB)
  char* base = (char*)d_ws; size_t off = 0;
  auto carve = [&](size_t bytes) -> char* {
    off = (off + 255) & ~(size_t)255;
    char* r = base + off; off += bytes; return r;
  };
  float* bufA   = (float*)carve(4096 * 256 * 4);
  float* bufB   = (float*)carve(4096 * 256 * 4);
  float* bufC   = (float*)carve(4096 * 256 * 4);
  float* bufAgg = (float*)carve(4096 * 256 * 4);
  _Float16* Xh  = (_Float16*)carve(4096 * 512 * 2);
  _Float16* Wh  = (_Float16*)carve(512 * 256 * 2);
  _Float16* Wh2 = (_Float16*)carve(256 * 256 * 2);
  float* teL0   = (float*)carve(64 * 256 * 4);
  float* te     = (float*)carve(64 * 256 * 4);
  _Float16* teh = (_Float16*)carve(64 * 256 * 2);
  float* cte    = (float*)carve(64 * 256 * 4);
  float* Amat   = (float*)carve(64 * 64 * 4);
  float* cntInv = (float*)carve(64 * 4);

  // time embedding: te = ffn(time)
  k_te0<<<64, 256, 0, stream>>>(tm, t0.g, t0.b, t0.mu, t0.var, t0.W, t0.bias, teL0);
  bnToH(teL0, t1, 0, teh, 64 * 256, 256, stream);
  cvtWT(t1.W, Wh, 256, stream);
  gemm(teh, Wh, t1.bias, nullptr, nullptr, te, 64, 256, 1, stream);

  // x = ffn(p)
  k_pre0<<<4096, 256, 0, stream>>>(p, pr0.g, pr0.b, pr0.mu, pr0.var, pr0.W, pr0.bias, bufB);
  bnToH(bufB, pr1, 0, Xh, 4096 * 256, 256, stream);
  cvtWT(pr1.W, Wh, 256, stream);
  gemm(Xh, Wh, pr1.bias, nullptr, nullptr, bufA, 4096, 256, 1, stream);

  // adjacency counts (deterministic integer atomics)
  k_buildA<<<1, 256, 0, stream>>>(edges, E, Amat, cntInv);

  // two graph convs with residuals
  graph_conv(c1p0, c1p1, c1u0, c1u1, bufA, bufB, bufC, bufAgg, bufC,
             te, teh, cte, Xh, Wh, Wh2, Amat, cntInv, stream);
  graph_conv(c2p0, c2p1, c2u0, c2u1, bufC, bufB, bufA, bufAgg, bufA,
             te, teh, cte, Xh, Wh, Wh2, Amat, cntInv, stream);

  // post ffn
  bnToH(bufA, po0, 0, Xh, 4096 * 256, 256, stream);
  cvtWT(po0.W, Wh, 256, stream);
  gemm(Xh, Wh, po0.bias, nullptr, nullptr, bufB, 4096, 256, 1, stream);
  bnToH(bufB, po1, 0, Xh, 4096 * 256, 256, stream);
  cvtWT(po1.W, Wh, 256, stream);
  gemm(Xh, Wh, po1.bias, nullptr, nullptr, bufC, 4096, 256, 1, stream);

  // final linear 256 -> 4
  k_final<<<64, 256, 0, stream>>>(bufC, Wl, bl, (float*)d_out);
}
